// PositiveWaveFunction_14439680049780
// MI455X (gfx1250) — matrix operand order repaired and checked by reading, not compiled, not run on hardware
//
#include <hip/hip_runtime.h>
#include <math.h>

typedef __attribute__((ext_vector_type(16))) _Float16 v16h;
typedef __attribute__((ext_vector_type(8)))  _Float16 v8h;
typedef __attribute__((ext_vector_type(8)))  float    v8f;

#if __has_builtin(__builtin_amdgcn_tanhf)
#define FAST_TANH(x) __builtin_amdgcn_tanhf(x)
#else
#define FAST_TANH(x) tanhf(x)
#endif

#define S_      1024
#define B_      256
#define H_      256
#define BT      16     // batch rows per workgroup (WMMA M)
#define NT      16     // 16-column N tiles covering H
#define THREADS 512    // 16 waves, one N tile per wave

// Single struct => guaranteed member order: hot arrays at low LDS offsets so
// every per-step DS access folds its address into the 16-bit immediate field.
struct Smem {
  _Float16 hs[BT * H_];              // 8 KB   resident hidden state [16][256]
  float    lws[2 * H_];              // 2 KB   lin_W rows
  float    xs[BT][2];                //        shifted input this step
  float    pp[BT * 32][2];           // 4 KB   logit partials
  float    lbs[2];
  // init-only, placed LAST (only touched when pinning B registers)
  alignas(32) _Float16 WB[NT * 8 * 32 * 16];   // 128 KB swizzled W_hh^T (WMMA B layout)
};

__global__ __launch_bounds__(THREADS, 1)
void rnn_fused_kernel(const float* __restrict__ data,
                      const float* __restrict__ W_ih,
                      const float* __restrict__ b_ih,
                      const float* __restrict__ W_hh,
                      const float* __restrict__ b_hh,
                      const float* __restrict__ lin_W,
                      const float* __restrict__ lin_b,
                      float* __restrict__ out) {
  __shared__ Smem sm;

  const int tid  = threadIdx.x;
  const int lane = tid & 31;
  const int wave = tid >> 5;           // 0..15; wave w owns N tile w
  const int wgB  = blockIdx.x * BT;    // global batch base

  // ---------------- one-time init ----------------
  for (int i = tid; i < BT * H_; i += THREADS) sm.hs[i] = (_Float16)0.0f;  // h0 = 0
  for (int i = tid; i < 2 * H_; i += THREADS)  sm.lws[i] = lin_W[i];
  if (tid < 2) sm.lbs[tid] = lin_b[tid];

  // Build WB: element (nt, kc, lane, j) = W_hh[n][K] with
  //   n = nt*16 + (lane&15),  K = kc*32 + (j>>3)*16 + (lane>>4)*8 + (j&7)
  for (int idx = tid; idx < NT * 8 * 32 * 16; idx += THREADS) {
    int j  = idx & 15;
    int ln = (idx >> 4) & 31;
    int kc = (idx >> 9) & 7;
    int nt = idx >> 12;
    int n  = nt * 16 + (ln & 15);
    int K  = kc * 32 + (j >> 3) * 16 + (ln >> 4) * 8 + (j & 7);
    sm.WB[idx] = (_Float16)W_hh[n * H_ + K];
  }

  // per-lane epilogue constants for this wave's N tile
  const int n0 = wave * 16 + (lane & 15);
  const float w00 = W_ih[n0 * 2 + 0], w01 = W_ih[n0 * 2 + 1];
  const float bs0 = b_ih[n0] + b_hh[n0];

  __syncthreads();

  // Pin this wave's B operands in registers: 8 x v16h = 64 VGPRs, pure f16 loads.
  v16h breg[8];
#pragma unroll
  for (int kc = 0; kc < 8; ++kc)
    breg[kc] = *(const v16h*)&sm.WB[((wave * 8 + kc) * 32 + lane) * 16];

  // Hoist loop-invariant LDS data into registers (compiler can't hoist LDS
  // loads across barriers itself): lin_W slice for this thread's phase-4 chunk
  // and the two lin_b scalars for phase 5.
  float lw0r[8], lw1r[8];
  {
    int c = tid & 31;
#pragma unroll
    for (int j = 0; j < 8; ++j) {
      lw0r[j] = sm.lws[c * 8 + j];
      lw1r[j] = sm.lws[H_ + c * 8 + j];
    }
  }
  const float lb0 = sm.lbs[0], lb1 = sm.lbs[1];

  // A-matrix LDS addressing (16-bit A 16x32 layout):
  // lane<16: row M=lane, K octets {0..7, 16..23}; lane>=16: {8..15, 24..31}
  const _Float16* abase = sm.hs + (lane & 15) * H_ + (lane >> 4) * 8;

  for (int s = 0; s < S_; ++s) {
    // -------- phase 1: stage shifted input x_t (x_0 = [1,0]) --------
    if (tid < 32) {
      int b = tid >> 1, i = tid & 1;
      float v;
      if (s == 0) v = (i == 0) ? 1.0f : 0.0f;
      else        v = data[(((size_t)(s - 1) * B_) + wgB + b) * 2 + i];
      sm.xs[b][i] = v;
      // prefetch next step's slice (global_prefetch_b8)
      if (s + 1 < S_)
        __builtin_prefetch(&data[(((size_t)s * B_) + wgB + b) * 2 + i], 0, 1);
    }

    // -------- phase 2: h @ W_hh^T, TWO independent WMMA chains --------
    v8f acc0 = {}; v8f acc1 = {};
#pragma unroll
    for (int kp = 0; kp < 4; ++kp) {
      const _Float16* ap0 = abase + (2 * kp)     * 32;
      const _Float16* ap1 = abase + (2 * kp + 1) * 32;
      v8h lo0 = *(const v8h*)(ap0);
      v8h hi0 = *(const v8h*)(ap0 + 16);
      v8h lo1 = *(const v8h*)(ap1);
      v8h hi1 = *(const v8h*)(ap1 + 16);
      v16h a0, a1;
#pragma unroll
      for (int e = 0; e < 8; ++e) {
        a0[e] = lo0[e]; a0[8 + e] = hi0[e];
        a1[e] = lo1[e]; a1[8 + e] = hi1[e];
      }
      acc0 = __builtin_amdgcn_wmma_f32_16x16x32_f16(false, a0, false, breg[2 * kp],
                                                    (short)0, acc0, false, false);
      acc1 = __builtin_amdgcn_wmma_f32_16x16x32_f16(false, a1, false, breg[2 * kp + 1],
                                                    (short)0, acc1, false, false);
    }

    __syncthreads();   // xs visible; all reads of old h complete

    // -------- phase 3: combine chains, input proj + bias, tanh, write h_new --------
    const int mg = (lane >> 4) * 8;    // C/D layout: VGPR r -> M = r + 8*(lane/16)
#pragma unroll
    for (int r = 0; r < 8; ++r) {
      int M = mg + r;
      float v0 = FAST_TANH((acc0[r] + acc1[r]) + sm.xs[M][0] * w00 + sm.xs[M][1] * w01 + bs0);
      sm.hs[M * H_ + n0] = (_Float16)v0;
    }

    __syncthreads();   // h_new visible to all waves

    // -------- phase 4: fused output projection partials --------
    {
      int b = tid >> 5, c = tid & 31;          // 32 chunks of 8 columns
      v8h h8 = *(const v8h*)(sm.hs + b * H_ + c * 8);
      float p0 = 0.f, p1 = 0.f;
#pragma unroll
      for (int j = 0; j < 8; ++j) {
        float f = (float)h8[j];
        p0 += f * lw0r[j];
        p1 += f * lw1r[j];
      }
      sm.pp[b * 32 + c][0] = p0;
      sm.pp[b * 32 + c][1] = p1;
    }

    __syncthreads();   // partials visible

    // -------- phase 5: reduce, 2-class softmax, store --------
    if (tid < BT) {
      float l0 = lb0, l1 = lb1;
#pragma unroll
      for (int c = 0; c < 32; ++c) {
        l0 += sm.pp[tid * 32 + c][0];
        l1 += sm.pp[tid * 32 + c][1];
      }
      float m  = fmaxf(l0, l1);
      float e0 = __expf(l0 - m), e1 = __expf(l1 - m);
      float inv = 1.0f / (e0 + e1);
      float* op = out + (((size_t)s * B_) + wgB + tid) * 2;
      op[0] = e0 * inv;
      op[1] = e1 * inv;
    }
  }
}

extern "C" void kernel_launch(void* const* d_in, const int* in_sizes, int n_in,
                              void* d_out, int out_size, void* d_ws, size_t ws_size,
                              hipStream_t stream) {
  const float* data  = (const float*)d_in[0];
  const float* W_ih  = (const float*)d_in[1];
  const float* b_ih  = (const float*)d_in[2];
  const float* W_hh  = (const float*)d_in[3];
  const float* b_hh  = (const float*)d_in[4];
  const float* lin_W = (const float*)d_in[5];
  const float* lin_b = (const float*)d_in[6];
  hipLaunchKernelGGL(rnn_fused_kernel, dim3(B_ / BT), dim3(THREADS), 0, stream,
                     data, W_ih, b_ih, W_hh, b_hh, lin_W, lin_b, (float*)d_out);
}
